// FBPINN_7164005450319
// MI455X (gfx1250) — matrix operand order repaired and checked by reading, not compiled, not run on hardware
//
#include <hip/hip_runtime.h>
#include <hip/hip_bf16.h>

typedef __attribute__((ext_vector_type(16))) _Float16 v16h;
typedef __attribute__((ext_vector_type(8)))  _Float16 v8h;
typedef __attribute__((ext_vector_type(8)))  float    v8f;

#define N_SUBD   16
#define WIDTH    128
#define NHID     3           // DEPTH-1 hidden 128x128 layers
#define HPAD     136         // padded f16 row stride (272 B: 16B-aligned, +4 dword bank rotate/row)
#define WAVES    8
#define MTILES   2           // 16-point M-tiles per wave
#define TILE_PTS (WAVES * MTILES * 16)   // 256 points per block

#define W_ELEMS   (NHID*WIDTH*HPAD)              // f16
#define HT_ELEMS  (WAVES*(MTILES*16)*HPAD)       // f16
#define F_ELEMS   (6*WIDTH)                      // f32: w0c,b0c,woutc,bhc[3]
#define SMEM_BYTES (W_ELEMS*2 + HT_ELEMS*2 + F_ELEMS*4)

__device__ __forceinline__ float fast_tanh(float x) {
#if __has_builtin(__builtin_amdgcn_tanhf)
  return __builtin_amdgcn_tanhf(x);
#else
  return tanhf(x);
#endif
}

__device__ __forceinline__ float fast_sigmoid(float z) {
  return 1.0f / (1.0f + __expf(-z));
}

__launch_bounds__(256)
__global__ void fbpinn_kernel(const float* __restrict__ x,
                              const float* __restrict__ W0,
                              const float* __restrict__ b0,
                              const float* __restrict__ Wh,
                              const float* __restrict__ bh,
                              const float* __restrict__ Wout,
                              const float* __restrict__ bout,
                              const float* __restrict__ centres,
                              const float* __restrict__ scales,
                              const float* __restrict__ mu_min,
                              const float* __restrict__ sd_min,
                              const float* __restrict__ mu_max,
                              const float* __restrict__ sd_max,
                              float* __restrict__ out, int npts)
{
  extern __shared__ char smem[];
  _Float16* Wlds  = (_Float16*)smem;                    // [NHID][WIDTH][HPAD]
  _Float16* htile = Wlds + W_ELEMS;                     // [WAVES][32][HPAD]
  float*    fbuf  = (float*)(htile + HT_ELEMS);
  float* w0c   = fbuf;            // [128]
  float* b0c   = fbuf + 128;      // [128]
  float* woutc = fbuf + 256;      // [128]
  float* bhc   = fbuf + 384;      // [3][128]

  const int tid  = threadIdx.x;
  const int wave = tid >> 5;
  const int lane = tid & 31;
  const int l16  = lane & 15;     // row-in-tile / N index
  const int hf   = lane >> 4;     // half-wave index

  _Float16* ht = htile + (wave * (MTILES * 16)) * HPAD; // this wave's 32xHPAD tile

  const int ptBase = blockIdx.x * TILE_PTS + wave * (MTILES * 16);
  float xp[MTILES];
  #pragma unroll
  for (int t = 0; t < MTILES; ++t)
    xp[t] = x[ptBase + t * 16 + l16];

  float outAcc[MTILES] = {0.0f, 0.0f};

  for (int s = 0; s < N_SUBD; ++s) {
    __syncthreads();   // previous iteration done reading Wlds/fbuf

    // ---- stage hidden weights (f32 -> f16), 8 elements per iteration ----
    // groups of 8 floats never cross a 128-wide row; LDS dst is 16B-aligned.
    for (int g = tid; g < (NHID * WIDTH * WIDTH) / 8; g += 256) {
      const int base = g * 8;
      const int l    = base >> 14;          // / 16384
      const int rem  = base & 16383;
      const float4* src =
          (const float4*)&Wh[(s * NHID + l) * (WIDTH * WIDTH) + rem];
      const float4 a = src[0];
      const float4 b = src[1];
      v8h pk;
      pk[0] = (_Float16)a.x; pk[1] = (_Float16)a.y;
      pk[2] = (_Float16)a.z; pk[3] = (_Float16)a.w;
      pk[4] = (_Float16)b.x; pk[5] = (_Float16)b.y;
      pk[6] = (_Float16)b.z; pk[7] = (_Float16)b.w;
      *(v8h*)&Wlds[(l * WIDTH + (rem >> 7)) * HPAD + (rem & 127)] = pk;
    }
    for (int idx = tid; idx < WIDTH; idx += 256) {
      w0c[idx]         = W0[s * WIDTH + idx];
      b0c[idx]         = b0[s * WIDTH + idx];
      woutc[idx]       = Wout[s * WIDTH + idx];
      bhc[0*WIDTH+idx] = bh[(s * NHID + 0) * WIDTH + idx];
      bhc[1*WIDTH+idx] = bh[(s * NHID + 1) * WIDTH + idx];
      bhc[2*WIDTH+idx] = bh[(s * NHID + 2) * WIDTH + idx];
    }
    __syncthreads();

    // prefetch next subdomain's hidden weights into cache while we compute
    if (s + 1 < N_SUBD) {
      const float* nw = Wh + ((s + 1) * NHID) * (WIDTH * WIDTH);
      #pragma unroll
      for (int rep = 0; rep < 3; ++rep) {
        const int off = (rep * 256 + tid) * 64;   // one probe per 256B line
        if (off < NHID * WIDTH * WIDTH) __builtin_prefetch(nw + off, 0, 0);
      }
    }

    // ---- layer 0: h0[p][w] = tanh(W0[w]*xi_p + b0[w]) ----
    const float sc     = fmaxf(scales[s], 1e-8f);
    const float centre = centres[s];
    #pragma unroll
    for (int t = 0; t < MTILES; ++t) {
      const float xi = (xp[t] - centre) / sc;
      const int   p  = t * 16 + l16;       // both half-waves: same point, split neurons
      #pragma unroll
      for (int wv = 0; wv < 64; wv += 8) {
        v8h pack;
        #pragma unroll
        for (int k = 0; k < 8; ++k) {
          const int w = 64 * hf + wv + k;
          pack[k] = (_Float16)fast_tanh(w0c[w] * xi + b0c[w]);
        }
        *(v8h*)(&ht[p * HPAD + 64 * hf + wv]) = pack;
      }
    }

    // ---- hidden layers: 128x128 GEMM per layer via WMMA, 2 M-tiles/wave ----
    for (int l = 0; l < NHID; ++l) {
      // A fragments for both tiles (register-resident across the whole layer)
      v16h afrag[MTILES][4];
      #pragma unroll
      for (int t = 0; t < MTILES; ++t) {
        #pragma unroll
        for (int c = 0; c < 4; ++c) {
          const _Float16* arow = &ht[(t * 16 + l16) * HPAD + c * 32 + hf * 8];
          v8h alo = *(const v8h*)(arow);
          v8h ahi = *(const v8h*)(arow + 16);
          afrag[t][c] = __builtin_shufflevector(alo, ahi,
                          0,1,2,3,4,5,6,7,8,9,10,11,12,13,14,15);
        }
      }

      // per N-block: zero-init C (inline SRC2=0), chain K, epilogue immediately
      #pragma unroll
      for (int j = 0; j < 8; ++j) {
        v8f acc0 = {}, acc1 = {};
        #pragma unroll
        for (int c = 0; c < 4; ++c) {
          const _Float16* brow =
              &Wlds[(l * WIDTH + 16 * j + l16) * HPAD + 32 * c + 8 * hf];
          v8h blo = *(const v8h*)(brow);
          v8h bhi = *(const v8h*)(brow + 16);
          v16h bfrag = __builtin_shufflevector(blo, bhi,
                         0,1,2,3,4,5,6,7,8,9,10,11,12,13,14,15);
          acc0 = __builtin_amdgcn_wmma_f32_16x16x32_f16(
                     false, afrag[0][c], false, bfrag, (short)0, acc0, false, false);
          acc1 = __builtin_amdgcn_wmma_f32_16x16x32_f16(
                     false, afrag[1][c], false, bfrag, (short)0, acc1, false, false);
        }
        // bias + tanh + writeback: element (M = r + 8*hf, N = 16j + l16)
        const float bb = bhc[l * WIDTH + 16 * j + l16];
        #pragma unroll
        for (int r = 0; r < 8; ++r) {
          ht[(r + 8 * hf) * HPAD + 16 * j + l16] =
              (_Float16)fast_tanh(acc0[r] + bb);
          ht[(16 + r + 8 * hf) * HPAD + 16 * j + l16] =
              (_Float16)fast_tanh(acc1[r] + bb);
        }
      }
    }

    // ---- output layer: raw_p = Wout . h_p + bout ----
    const float bo = bout[s];
    const float mmin = mu_min[s], smin = sd_min[s];
    const float mmax = mu_max[s], smax = sd_max[s];
    #pragma unroll
    for (int t = 0; t < MTILES; ++t) {
      float dsum = 0.0f;
      #pragma unroll
      for (int cb = 0; cb < 64; cb += 8) {
        v8h hv = *(const v8h*)(&ht[(t * 16 + l16) * HPAD + 64 * hf + cb]);
        #pragma unroll
        for (int k = 0; k < 8; ++k)
          dsum += woutc[64 * hf + cb + k] * (float)hv[k];
      }
      dsum += __shfl_xor(dsum, 16, 32);   // combine the two half-wave partial dots
      const float raw = dsum + bo;
      const float wlo = fast_sigmoid((xp[t] - mmin) / smin);
      const float whi = fast_sigmoid((mmax - xp[t]) / smax);
      outAcc[t] += wlo * whi * raw;
    }
  }

  if (hf == 0) {
    #pragma unroll
    for (int t = 0; t < MTILES; ++t)
      out[ptBase + t * 16 + l16] = outAcc[t];
  }
}

extern "C" void kernel_launch(void* const* d_in, const int* in_sizes, int n_in,
                              void* d_out, int out_size, void* d_ws, size_t ws_size,
                              hipStream_t stream) {
  (void)n_in; (void)out_size; (void)d_ws; (void)ws_size;
  const float* x       = (const float*)d_in[0];
  const float* W0      = (const float*)d_in[1];
  const float* b0      = (const float*)d_in[2];
  const float* Wh      = (const float*)d_in[3];
  const float* bh      = (const float*)d_in[4];
  const float* Wout    = (const float*)d_in[5];
  const float* bout    = (const float*)d_in[6];
  const float* centres = (const float*)d_in[7];
  const float* scales  = (const float*)d_in[8];
  const float* mu_min  = (const float*)d_in[9];
  const float* sd_min  = (const float*)d_in[10];
  const float* mu_max  = (const float*)d_in[11];
  const float* sd_max  = (const float*)d_in[12];

  const int npts = in_sizes[0];           // 65536, multiple of TILE_PTS
  const int grid = npts / TILE_PTS;

  static_assert(SMEM_BYTES < 320 * 1024, "LDS over WGP budget");
  hipFuncSetAttribute((const void*)fbpinn_kernel,
                      hipFuncAttributeMaxDynamicSharedMemorySize, SMEM_BYTES);

  fbpinn_kernel<<<grid, 256, SMEM_BYTES, stream>>>(
      x, W0, b0, Wh, bh, Wout, bout, centres, scales,
      mu_min, sd_min, mu_max, sd_max, (float*)d_out, npts);
}
